// CausalSelfAttention_34119220199749
// MI455X (gfx1250) — matrix operand order repaired and checked by reading, not compiled, not run on hardware
//
#include <hip/hip_runtime.h>

// ---------------------------------------------------------------------------
// Causal self-attention for MI455X (gfx1250, wave32, WMMA).
// Pipeline: f32->bf16 convert | QKV GEMM + fused RoPE | flash attention | proj.
// All matmuls via v_wmma_f32_16x16x32_bf16 (f32 accumulation).
// GEMM waves compute M=32 tiles so each B fragment feeds two WMMAs (halves
// L2 traffic and wait exposure). TDM prefetches the first K tile per block.
// ---------------------------------------------------------------------------

typedef __bf16 bf16;
typedef __attribute__((ext_vector_type(16))) __bf16 v16bf;
typedef __attribute__((ext_vector_type(8)))  __bf16 v8bf;
typedef __attribute__((ext_vector_type(8)))  float  v8f;

#if defined(__AMDGCN__) && __has_builtin(__builtin_amdgcn_tensor_load_to_lds)
#define HAVE_TDM 1
typedef __attribute__((ext_vector_type(4))) unsigned int v4u;
typedef __attribute__((ext_vector_type(8))) int v8i;
typedef __attribute__((ext_vector_type(4))) int v4i;
#endif

#define DEV static __device__ __forceinline__

constexpr int Bc = 2;
constexpr int Tc = 2048;
constexpr int Dc = 2048;
constexpr int Hc = 16;
constexpr int DHc = 128;

// round-to-nearest-even f32 -> bf16 (as raw u16)
DEV unsigned short f2bf(float f) {
  unsigned int u = __float_as_uint(f);
  u += 0x7FFFu + ((u >> 16) & 1u);
  return (unsigned short)(u >> 16);
}

DEV v16bf ld16(const unsigned short* p) {
  return *reinterpret_cast<const v16bf*>(p);
}
DEV v8bf ld8(const unsigned short* p) {
  return *reinterpret_cast<const v8bf*>(p);
}
DEV v16bf combine(v8bf a, v8bf b) {
  v16bf r;
#pragma unroll
  for (int i = 0; i < 8; ++i) { r[i] = a[i]; r[i + 8] = b[i]; }
  return r;
}
DEV v8f vzero() {
  v8f z;
#pragma unroll
  for (int i = 0; i < 8; ++i) z[i] = 0.0f;
  return z;
}
DEV v8f wmma_bf16(v16bf a, v16bf b, v8f c) {
  // D = A(16x32) * B(32x16) + C, f32 accumulate
  return __builtin_amdgcn_wmma_f32_16x16x32_bf16(
      false, a, false, b, (short)0, c, false, false);
}

// A fragment loader: lane lo holds row, 16 contiguous K elements at K offset
// hi*16 (matches contiguous B fragments).
DEV v16bf ldA(const unsigned short* row, int kk, int hi) {
  return combine(ld8(row + kk + hi * 16), ld8(row + kk + hi * 16 + 8));
}

// ---------------------------------------------------------------------------
// Kernel 0: f32 -> bf16 conversion
// ---------------------------------------------------------------------------
__global__ void cvt_f32_bf16(const float* __restrict__ in,
                             unsigned short* __restrict__ out, int n) {
  int i = blockIdx.x * blockDim.x + threadIdx.x;
  if (i < n) out[i] = f2bf(in[i]);
}

// ---------------------------------------------------------------------------
// Kernel 1: QKV GEMM (x @ Wqkv^T) with fused RoPE on Q,K. V stored transposed.
//   xb:  [B*T, D] bf16        wq: [3D, D] bf16
//   q,k: [B,H,T,Dh] bf16      vt: [B,H,Dh,T] bf16
// One wave computes a 32(row) x 128(one head) tile: 2 A fragments x 8 B
// fragments = 16 f32 accumulators; each B fragment is reused by 2 WMMAs.
// ---------------------------------------------------------------------------
__global__ void __launch_bounds__(256) qkv_rope_kernel(
    const unsigned short* __restrict__ xb,
    const unsigned short* __restrict__ wq,
    unsigned short* __restrict__ q,
    unsigned short* __restrict__ k,
    unsigned short* __restrict__ vt) {
  const int lane = threadIdx.x & 31;
  const int wave = threadIdx.x >> 5;
  const int tile = blockIdx.x * 8 + wave;   // 128 row tiles * 48 head groups
  const int rowTile = tile / 48;            // 0..127
  const int g = tile % 48;                  // 0..15 Q heads, 16..31 K, 32..47 V
  const int m0 = rowTile * 32;              // row in [0, B*T)
  const int lo = lane & 15, hi = lane >> 4;
  const int colBase = g * DHc;              // column in [0, 3D)

  const unsigned short* arow0 = xb + (size_t)(m0 + lo) * Dc;
  const unsigned short* arow1 = xb + (size_t)(m0 + 16 + lo) * Dc;
  const unsigned short* brow[8];
#pragma unroll
  for (int n = 0; n < 8; ++n)
    brow[n] = wq + (size_t)(colBase + 16 * n + lo) * Dc + hi * 16;

  v8f acc[2][8];
#pragma unroll
  for (int r = 0; r < 2; ++r)
#pragma unroll
    for (int n = 0; n < 8; ++n) acc[r][n] = vzero();

  for (int kk = 0; kk < Dc; kk += 32) {
    __builtin_prefetch(arow0 + kk + 256, 0, 3);
    __builtin_prefetch(arow1 + kk + 256, 0, 3);
    v16bf a0 = ldA(arow0, kk, hi);
    v16bf a1 = ldA(arow1, kk, hi);
#pragma unroll
    for (int n = 0; n < 8; ++n) {
      v16bf bf = ld16(brow[n] + kk);
      acc[0][n] = wmma_bf16(a0, bf, acc[0][n]);
      acc[1][n] = wmma_bf16(a1, bf, acc[1][n]);
    }
  }

  const int b = m0 / Tc;   // all 32 rows share the same batch (T % 32 == 0)
  const int t0 = m0 % Tc;

  if (g < 32) {
    // Q or K: apply RoPE. acc n covers dims d = 16n + lo; pair d <-> d +/- 64
    // lives in acc n +/- 4 at the same lane/element -> register-local rotate.
    const int h = (g < 16) ? g : (g - 16);
    unsigned short* dst =
        ((g < 16) ? q : k) + (size_t)((b * Hc + h) * Tc) * DHc;
    float invf[4];
#pragma unroll
    for (int i = 0; i < 4; ++i) {
      float fi = (float)(16 * i + lo);               // freq index = d & 63
      invf[i] = __expf(fi * (-0.14391565f));         // -ln(10000)/64
    }
#pragma unroll
    for (int r = 0; r < 2; ++r) {
#pragma unroll
      for (int n = 0; n < 8; ++n) {
#pragma unroll
        for (int j = 0; j < 8; ++j) {
          int row = 16 * r + j + 8 * hi;             // 0..31
          float t = (float)(t0 + row);
          float ang = t * invf[n & 3];
          float c = __cosf(ang), s = __sinf(ang);
          float x = acc[r][n][j];
          float rv = (n < 4) ? -acc[r][n + 4][j] : acc[r][n - 4][j];
          int d = 16 * n + lo;
          dst[(size_t)(t0 + row) * DHc + d] = f2bf(x * c + rv * s);
        }
      }
    }
  } else {
    // V: store transposed [B,H,Dh,T] so PV B-fragments load contiguously.
    const int h = g - 32;
    unsigned short* dst = vt + (size_t)((b * Hc + h) * DHc) * Tc;
#pragma unroll
    for (int r = 0; r < 2; ++r) {
#pragma unroll
      for (int n = 0; n < 8; ++n) {
#pragma unroll
        for (int j = 0; j < 8; ++j) {
          int d = 16 * n + lo;
          int row = 16 * r + j + 8 * hi;
          dst[(size_t)d * Tc + (t0 + row)] = f2bf(acc[r][n][j]);
        }
      }
    }
  }
}

// ---------------------------------------------------------------------------
// Kernel 2: flash attention, one wave (32 threads) per (b, h, 16-query tile).
//   q,k: [B,H,T,Dh] bf16   vt: [B,H,Dh,T] bf16   o: [B,T,D] bf16
// Per 32-key tile: 8 WMMA for S = QK^T, online softmax (shfl row-reduce),
// P rearranged C-layout -> A-layout through LDS, 8 WMMA for O += P V.
// V fragments are prefetched before the softmax VALU block so exp/shfl work
// covers their latency. TDM prefetches the first K tile into LDS/GL2.
// ---------------------------------------------------------------------------
__global__ void __launch_bounds__(32) attn_kernel(
    const unsigned short* __restrict__ q,
    const unsigned short* __restrict__ k,
    const unsigned short* __restrict__ vt,
    unsigned short* __restrict__ o) {
  __shared__ unsigned short p_lds[16 * 32];

  const int lane = threadIdx.x & 31;
  const int lo = lane & 15, hi = lane >> 4;
  const int bid = blockIdx.x;          // b*H*(T/16) blocks
  const int qt = bid & 127;            // T/16 = 128
  const int bh = bid >> 7;             // 0..31 = b*H + h
  const int qBase = qt * 16;

  const unsigned short* qbase = q + (size_t)bh * Tc * DHc;
  const unsigned short* kbase = k + (size_t)bh * Tc * DHc;
  const unsigned short* vbase = vt + (size_t)bh * DHc * Tc;

#ifdef HAVE_TDM
  // TDM prefetch of the first 32x128 K tile (bf16) into an LDS landing zone.
  // Never read back: acts as a DMA prefetch that warms GL2/LDS while the wave
  // sets up Q fragments. Descriptor per CDNA5 ISA 8.3-8.4.
  __shared__ unsigned short tdm_buf[32 * DHc];
  {
    unsigned lds_off = (unsigned)(size_t)(void*)&tdm_buf[0];
    unsigned long long ga = (unsigned long long)(size_t)kbase;
    v4u g0 = {
        1u,                                     // count=1, user descriptor
        lds_off,                                // lds_addr (bytes)
        (unsigned)(ga & 0xFFFFFFFFu),           // global_addr[31:0]
        (unsigned)(ga >> 32) | (2u << 30)       // global_addr[56:32] | type=2
    };
    v8i g1 = {
        (int)(1u << 16),          // workgroup_mask=0, data_size=1 (2 bytes)
        (int)((unsigned)DHc << 16),  // tensor_dim0[15:0]=128 (bits 63:48)
        (int)((unsigned)Tc << 16),   // dim0 hi=0 | tensor_dim1[15:0]=2048
        (int)((unsigned)DHc << 16),  // dim1 hi=0 | tile_dim0=128
        32,                          // tile_dim1=32, tile_dim2=0
        DHc,                         // tensor_dim0_stride=128
        0, 0                         // stride hi, tensor_dim1_stride=0 (2D)
    };
    v4i g2 = {0, 0, 0, 0};
    v4i g3 = {0, 0, 0, 0};
#if __clang_major__ >= 23
    v8i g4 = {0, 0, 0, 0, 0, 0, 0, 0};
    __builtin_amdgcn_tensor_load_to_lds(g0, g1, g2, g3, g4, 0);
#else
    __builtin_amdgcn_tensor_load_to_lds(g0, g1, g2, g3, 0);
#endif
  }
#endif

  // Q fragments for the 4 K-steps of Dh=128 (kept in registers).
  v16bf qa[4];
  {
    const unsigned short* qrow = qbase + (size_t)(qBase + lo) * DHc;
#pragma unroll
    for (int s = 0; s < 4; ++s) qa[s] = ldA(qrow, 32 * s, hi);
  }

  v8f oacc[8];
#pragma unroll
  for (int n = 0; n < 8; ++n) oacc[n] = vzero();
  float m[8], l[8];
#pragma unroll
  for (int j = 0; j < 8; ++j) { m[j] = -3.0e38f; l[j] = 0.0f; }

  const float scale = 0.08838834764831845f;   // 1/sqrt(128)

  for (int jj = 0; jj < qBase + 16; jj += 32) {
    // ---- preload all 8 K fragments, then S = Q K^T ----------------------
    v16bf kb[8];
#pragma unroll
    for (int ks = 0; ks < 4; ++ks) {
      kb[ks] = ld16(kbase + (size_t)(jj + lo) * DHc + 32 * ks + hi * 16);
      kb[4 + ks] =
          ld16(kbase + (size_t)(jj + 16 + lo) * DHc + 32 * ks + hi * 16);
    }
    v8f s0 = vzero(), s1 = vzero();
#pragma unroll
    for (int ks = 0; ks < 4; ++ks) {
      s0 = wmma_bf16(qa[ks], kb[ks], s0);
      s1 = wmma_bf16(qa[ks], kb[4 + ks], s1);
    }

    // ---- prefetch V fragments; softmax VALU work covers their latency ---
    v16bf vb[8];
#pragma unroll
    for (int n = 0; n < 8; ++n)
      vb[n] = ld16(vbase + (size_t)(16 * n + lo) * Tc + jj + hi * 16);

    // ---- scale + causal mask + online softmax ---------------------------
    float p0[8], p1[8];
#pragma unroll
    for (int j = 0; j < 8; ++j) {
      int tq = qBase + j + 8 * hi;
      float v0 = s0[j] * scale;
      float v1 = s1[j] * scale;
      if (jj + lo > tq) v0 = -3.0e38f;
      if (jj + 16 + lo > tq) v1 = -3.0e38f;
      float mx = fmaxf(v0, v1);
#pragma unroll
      for (int d = 1; d < 16; d <<= 1) mx = fmaxf(mx, __shfl_xor(mx, d, 32));
      float mn = fmaxf(m[j], mx);
      float a = __expf(m[j] - mn);
      p0[j] = __expf(v0 - mn);
      p1[j] = __expf(v1 - mn);
      float sum = p0[j] + p1[j];
#pragma unroll
      for (int d = 1; d < 16; d <<= 1) sum += __shfl_xor(sum, d, 32);
      l[j] = l[j] * a + sum;
      m[j] = mn;
#pragma unroll
      for (int n = 0; n < 8; ++n) oacc[n][j] *= a;
    }

    // ---- P: C-layout -> A-layout through LDS (single wave, DS in-order) --
#pragma unroll
    for (int j = 0; j < 8; ++j) {
      int r = j + 8 * hi;
      p_lds[r * 32 + lo] = f2bf(p0[j]);
      p_lds[r * 32 + 16 + lo] = f2bf(p1[j]);
    }
    asm volatile("" ::: "memory");
    v16bf pa;
    {
      const unsigned short* pr = p_lds + lo * 32;
      pa = combine(ld8(pr + hi * 16), ld8(pr + hi * 16 + 8));
    }
    asm volatile("" ::: "memory");

    // ---- O += P V (V^T rows are contiguous over key index) --------------
#pragma unroll
    for (int n = 0; n < 8; ++n) oacc[n] = wmma_bf16(pa, vb[n], oacc[n]);
  }

#if defined(HAVE_TDM) && __has_builtin(__builtin_amdgcn_s_wait_tensorcnt)
  __builtin_amdgcn_s_wait_tensorcnt(0);   // drain the TDM prefetch
#endif

  // ---- normalize and store O into [B,T,D] -------------------------------
  float inv[8];
#pragma unroll
  for (int j = 0; j < 8; ++j) inv[j] = 1.0f / l[j];
  const int b = bh >> 4, h = bh & 15;
#pragma unroll
  for (int n = 0; n < 8; ++n) {
#pragma unroll
    for (int j = 0; j < 8; ++j) {
      int t = qBase + j + 8 * hi;
      int col = h * DHc + 16 * n + lo;
      o[(size_t)(b * Tc + t) * Dc + col] = f2bf(oacc[n][j] * inv[j]);
    }
  }
}

// ---------------------------------------------------------------------------
// Kernel 3: output projection  out = O @ Wproj^T  (f32 result)
// One wave computes a 32 x 64 tile: 2 A fragments x 4 B fragments = 8 accs;
// each B fragment is reused by 2 WMMAs.
// ---------------------------------------------------------------------------
__global__ void __launch_bounds__(256) proj_kernel(
    const unsigned short* __restrict__ ob,
    const unsigned short* __restrict__ wp,
    float* __restrict__ out) {
  const int lane = threadIdx.x & 31;
  const int wave = threadIdx.x >> 5;
  const int tile = blockIdx.x * 8 + wave;   // 128 row tiles * 32 col tiles
  const int rowTile = tile >> 5;
  const int nt = tile & 31;
  const int m0 = rowTile * 32, n0 = nt * 64;
  const int lo = lane & 15, hi = lane >> 4;

  const unsigned short* arow0 = ob + (size_t)(m0 + lo) * Dc;
  const unsigned short* arow1 = ob + (size_t)(m0 + 16 + lo) * Dc;
  const unsigned short* brow[4];
#pragma unroll
  for (int c = 0; c < 4; ++c)
    brow[c] = wp + (size_t)(n0 + 16 * c + lo) * Dc + hi * 16;

  v8f acc[2][4];
#pragma unroll
  for (int r = 0; r < 2; ++r)
#pragma unroll
    for (int c = 0; c < 4; ++c) acc[r][c] = vzero();

  for (int kk = 0; kk < Dc; kk += 32) {
    __builtin_prefetch(arow0 + kk + 256, 0, 3);
    __builtin_prefetch(arow1 + kk + 256, 0, 3);
    v16bf a0 = ldA(arow0, kk, hi);
    v16bf a1 = ldA(arow1, kk, hi);
#pragma unroll
    for (int c = 0; c < 4; ++c) {
      v16bf bf = ld16(brow[c] + kk);
      acc[0][c] = wmma_bf16(a0, bf, acc[0][c]);
      acc[1][c] = wmma_bf16(a1, bf, acc[1][c]);
    }
  }

#pragma unroll
  for (int r = 0; r < 2; ++r) {
#pragma unroll
    for (int c = 0; c < 4; ++c) {
#pragma unroll
      for (int j = 0; j < 8; ++j) {
        int row = m0 + 16 * r + j + 8 * hi;
        int col = n0 + 16 * c + lo;
        out[(size_t)row * Dc + col] = acc[r][c][j];
      }
    }
  }
}

// ---------------------------------------------------------------------------
// Host launcher
// ---------------------------------------------------------------------------
extern "C" void kernel_launch(void* const* d_in, const int* in_sizes, int n_in,
                              void* d_out, int out_size, void* d_ws,
                              size_t ws_size, hipStream_t stream) {
  (void)in_sizes; (void)n_in; (void)out_size; (void)ws_size;

  const float* x = (const float*)d_in[0];        // [B,T,D]
  const float* wqkv = (const float*)d_in[1];     // [3D,D]
  const float* wproj = (const float*)d_in[2];    // [D,D]
  float* out = (float*)d_out;                    // [B,T,D]
  unsigned short* ws = (unsigned short*)d_ws;

  // workspace layout (bf16 elements, all 32B aligned)
  const size_t N_XB = (size_t)Bc * Tc * Dc;      // 8388608
  const size_t N_WQ = (size_t)3 * Dc * Dc;       // 12582912
  const size_t N_WP = (size_t)Dc * Dc;           // 4194304
  const size_t N_BHTD = (size_t)Bc * Hc * Tc * DHc;  // 8388608

  size_t off = 0;
  unsigned short* XB = ws + off; off += N_XB;
  unsigned short* WQ = ws + off; off += N_WQ;
  unsigned short* WP = ws + off; off += N_WP;
  unsigned short* Q  = ws + off; off += N_BHTD;
  unsigned short* K  = ws + off; off += N_BHTD;
  unsigned short* VT = ws + off; off += N_BHTD;
  unsigned short* O  = ws + off; off += N_BHTD;   // total ~117.4 MB

  cvt_f32_bf16<<<(int)((N_XB + 255) / 256), 256, 0, stream>>>(x, XB, (int)N_XB);
  cvt_f32_bf16<<<(int)((N_WQ + 255) / 256), 256, 0, stream>>>(wqkv, WQ, (int)N_WQ);
  cvt_f32_bf16<<<(int)((N_WP + 255) / 256), 256, 0, stream>>>(wproj, WP, (int)N_WP);

  // 128 row tiles (32 rows) * 48 head groups = 6144 wave-tiles, 8 waves/block
  qkv_rope_kernel<<<768, 256, 0, stream>>>(XB, WQ, Q, K, VT);

  // B*H*(T/16) = 4096 blocks, one wave each
  attn_kernel<<<4096, 32, 0, stream>>>(Q, K, VT, O);

  // 128 row tiles (32 rows) * 32 col tiles = 4096 wave-tiles, 8 waves/block
  proj_kernel<<<512, 256, 0, stream>>>(O, WP, out);
}
